// SingleHeadAttention_86595130622603
// MI455X (gfx1250) — compile-verified, matrix operand order checked by reading, not run on hardware
//
#include <hip/hip_runtime.h>
#include <hip/hip_bf16.h>
#include <math.h>

typedef __bf16 bf16_t;
typedef __attribute__((ext_vector_type(16))) __bf16 v16bf;
typedef __attribute__((ext_vector_type(8)))  __bf16 v8bf;
typedef __attribute__((ext_vector_type(8)))  float  v8f;

#define EMBED   512
#define SEQ_P   2048
#define NBATCH  4
#define M_TOTAL (NBATCH * SEQ_P)   /* 8192 */
#define QKV_D   (3 * EMBED)        /* 1536 */
#define SCALING 0.04419417382415922f  /* 512^-0.5 */

union FragU { v16bf f; v8bf h[2]; };

// 16-bit A/B fragment for v_wmma_*_16x16x32: per ISA 7.12.2, lane (l%16) owns
// row/col, half = l/16 selects K = half*8 + [0..7] (elems 0..7) and
// K = 16 + half*8 + [0..7] (elems 8..15): two contiguous 16B chunks.
__device__ __forceinline__ v16bf load_frag(const bf16_t* __restrict__ base,
                                           int ld, int row, int k0, int lane)
{
    int r    = row + (lane & 15);
    int half = lane >> 4;
    const bf16_t* p = base + (size_t)r * ld + (k0 + half * 8);
    FragU u;
    u.h[0] = *(const v8bf*)(p);
    u.h[1] = *(const v8bf*)(p + 16);
    return u.f;
}

__device__ __forceinline__ v8f wmma_bf16(v16bf a, v16bf b, v8f c)
{
    return __builtin_amdgcn_wmma_f32_16x16x32_bf16(false, a, false, b,
                                                   (short)0, c, false, false);
}

/* ---------------- kernel 1: fp32 -> bf16 conversion ---------------- */
__global__ void cvt_f32_bf16(const float* __restrict__ s, bf16_t* __restrict__ d, int n)
{
    int i      = blockIdx.x * blockDim.x + threadIdx.x;
    int stride = gridDim.x * blockDim.x;
    for (; i < n; i += stride) d[i] = (bf16_t)s[i];
}

/* ---------------- kernel 2: QKV projection GEMM --------------------
 * D = X(8192x512) * Wqkv^T(512x1536) + b ; per-wave 32x64 tile
 * (2 A-frags x 4 B-frags -> 8 WMMA per 12 b128 loads).
 * Q scaled, K plain, V written transposed into Vt[n][c][p].          */
__global__ __launch_bounds__(128) void qkv_gemm(
    const bf16_t* __restrict__ xb, const bf16_t* __restrict__ wqb,
    const float* __restrict__ b_qkv,
    bf16_t* __restrict__ Qs, bf16_t* __restrict__ Kb, bf16_t* __restrict__ Vt)
{
    int lane = threadIdx.x & 31;
    int wv   = threadIdx.x >> 5;
    int half = lane >> 4;
    int r0   = blockIdx.y * 128 + wv * 32;
    int c0   = blockIdx.x * 64;

    v8f acc[2][4] = {};
    for (int k = 0; k < EMBED; k += 32) {
        v16bf a0 = load_frag(xb, EMBED, r0,      k, lane);
        v16bf a1 = load_frag(xb, EMBED, r0 + 16, k, lane);
        if (k + 32 < EMBED) {
            __builtin_prefetch(xb + (size_t)(r0 + (lane & 15)) * EMBED + k + 32, 0, 1);
        }
#pragma unroll
        for (int t = 0; t < 4; ++t) {
            v16bf b = load_frag(wqb, EMBED, c0 + t * 16, k, lane);
            acc[0][t] = wmma_bf16(a0, b, acc[0][t]);
            acc[1][t] = wmma_bf16(a1, b, acc[1][t]);
        }
    }

    int nn = lane & 15;
#pragma unroll
    for (int t = 0; t < 4; ++t) {
        int   col  = c0 + t * 16 + nn;
        float bias = b_qkv[col];
#pragma unroll
        for (int rt = 0; rt < 2; ++rt) {
#pragma unroll
            for (int v = 0; v < 8; ++v) {
                int   row = r0 + rt * 16 + half * 8 + v;
                float val = acc[rt][t][v] + bias;
                if (col < EMBED) {
                    Qs[(size_t)row * EMBED + col] = (bf16_t)(val * SCALING);
                } else if (col < 2 * EMBED) {
                    Kb[(size_t)row * EMBED + (col - EMBED)] = (bf16_t)val;
                } else {
                    int nb = row >> 11;            /* row / 2048 */
                    int p  = row & (SEQ_P - 1);
                    int c  = col - 2 * EMBED;
                    Vt[((size_t)nb * EMBED + c) * SEQ_P + p] = (bf16_t)val;
                }
            }
        }
    }
}

/* ---------------- kernel 3: scores + softmax (fused) ---------------
 * One 256-thread block per (batch, 16 query rows). Each of the 8 waves
 * owns 256 key columns (16 WMMA accumulators). Softmax is done with
 * wave shfl_xor reductions + tiny LDS cross-wave reduction.           */
__global__ __launch_bounds__(256) void attn_scores_softmax(
    const bf16_t* __restrict__ Qs, const bf16_t* __restrict__ Kb,
    bf16_t* __restrict__ Pb)
{
    __shared__ float red[8 * 16];
    __shared__ float rowMax[16];
    __shared__ float rowSum[16];

    int lane = threadIdx.x & 31;
    int wv   = threadIdx.x >> 5;    /* 0..7 */
    int half = lane >> 4;
    int nb   = blockIdx.z;
    int p0   = blockIdx.x * 16;

    const bf16_t* Qn = Qs + (size_t)nb * SEQ_P * EMBED;
    const bf16_t* Kn = Kb + (size_t)nb * SEQ_P * EMBED;
    bf16_t*       Pn = Pb + (size_t)nb * SEQ_P * SEQ_P;

    int cw = wv * 256;
    v8f acc[16] = {};
    for (int k = 0; k < EMBED; k += 32) {
        v16bf a = load_frag(Qn, EMBED, p0, k, lane);
#pragma unroll
        for (int t = 0; t < 16; ++t) {
            v16bf b = load_frag(Kn, EMBED, cw + t * 16, k, lane);
            acc[t] = wmma_bf16(a, b, acc[t]);
        }
    }

    /* per-row max: across tiles, then across the 16 lanes of each half */
    float pmax[8];
#pragma unroll
    for (int v = 0; v < 8; ++v) {
        float m = acc[0][v];
#pragma unroll
        for (int t = 1; t < 16; ++t) m = fmaxf(m, acc[t][v]);
#pragma unroll
        for (int mk = 1; mk < 16; mk <<= 1) m = fmaxf(m, __shfl_xor(m, mk, 32));
        pmax[v] = m;
    }
    if ((lane & 15) == 0) {
#pragma unroll
        for (int v = 0; v < 8; ++v) red[wv * 16 + half * 8 + v] = pmax[v];
    }
    __syncthreads();
    if (threadIdx.x < 16) {
        float m = red[threadIdx.x];
        for (int w = 1; w < 8; ++w) m = fmaxf(m, red[w * 16 + threadIdx.x]);
        rowMax[threadIdx.x] = m;
    }
    __syncthreads();

    float smax[8], psum[8];
#pragma unroll
    for (int v = 0; v < 8; ++v) { smax[v] = rowMax[half * 8 + v]; psum[v] = 0.f; }
#pragma unroll
    for (int t = 0; t < 16; ++t) {
#pragma unroll
        for (int v = 0; v < 8; ++v) {
            float e = __expf(acc[t][v] - smax[v]);
            acc[t][v] = e;
            psum[v] += e;
        }
    }
#pragma unroll
    for (int v = 0; v < 8; ++v) {
        float s = psum[v];
#pragma unroll
        for (int mk = 1; mk < 16; mk <<= 1) s += __shfl_xor(s, mk, 32);
        psum[v] = s;
    }
    __syncthreads();             /* red is free again */
    if ((lane & 15) == 0) {
#pragma unroll
        for (int v = 0; v < 8; ++v) red[wv * 16 + half * 8 + v] = psum[v];
    }
    __syncthreads();
    if (threadIdx.x < 16) {
        float s = red[threadIdx.x];
        for (int w = 1; w < 8; ++w) s += red[w * 16 + threadIdx.x];
        rowSum[threadIdx.x] = s;
    }
    __syncthreads();

    float inv[8];
#pragma unroll
    for (int v = 0; v < 8; ++v) inv[v] = 1.0f / rowSum[half * 8 + v];

    int nn = lane & 15;
#pragma unroll
    for (int t = 0; t < 16; ++t) {
        int col = cw + t * 16 + nn;
#pragma unroll
        for (int v = 0; v < 8; ++v) {
            int row = p0 + half * 8 + v;
            Pn[(size_t)row * SEQ_P + col] = (bf16_t)(acc[t][v] * inv[v]);
        }
    }
}

/* ---------------- kernel 4: O = P @ V (via Vt), 32x64 per wave ----- */
__global__ __launch_bounds__(128) void av_gemm(
    const bf16_t* __restrict__ Pb, const bf16_t* __restrict__ Vt,
    bf16_t* __restrict__ Ob)
{
    int lane = threadIdx.x & 31;
    int wv   = threadIdx.x >> 5;
    int half = lane >> 4;
    int nb   = blockIdx.z;
    int r0   = blockIdx.y * 128 + wv * 32;
    int c0   = blockIdx.x * 64;

    const bf16_t* Pn = Pb + (size_t)nb * SEQ_P * SEQ_P;
    const bf16_t* Vn = Vt + (size_t)nb * EMBED * SEQ_P;

    v8f acc[2][4] = {};
    for (int k = 0; k < SEQ_P; k += 32) {
        v16bf a0 = load_frag(Pn, SEQ_P, r0,      k, lane);
        v16bf a1 = load_frag(Pn, SEQ_P, r0 + 16, k, lane);
        if (k + 32 < SEQ_P) {
            __builtin_prefetch(Pn + (size_t)(r0 + (lane & 15)) * SEQ_P + k + 32, 0, 1);
        }
#pragma unroll
        for (int t = 0; t < 4; ++t) {
            v16bf b = load_frag(Vn, SEQ_P, c0 + t * 16, k, lane);
            acc[0][t] = wmma_bf16(a0, b, acc[0][t]);
            acc[1][t] = wmma_bf16(a1, b, acc[1][t]);
        }
    }

    int nn = lane & 15;
#pragma unroll
    for (int t = 0; t < 4; ++t) {
        int col = c0 + t * 16 + nn;
#pragma unroll
        for (int rt = 0; rt < 2; ++rt) {
#pragma unroll
            for (int v = 0; v < 8; ++v) {
                int row = r0 + rt * 16 + half * 8 + v;
                Ob[(((size_t)nb * SEQ_P) + row) * EMBED + col] = (bf16_t)acc[rt][t][v];
            }
        }
    }
}

/* ---------------- kernel 5: output projection + bias, 32x64 -------- */
__global__ __launch_bounds__(128) void out_gemm(
    const bf16_t* __restrict__ Ob, const bf16_t* __restrict__ wob,
    const float* __restrict__ b_out, float* __restrict__ out)
{
    int lane = threadIdx.x & 31;
    int wv   = threadIdx.x >> 5;
    int half = lane >> 4;
    int r0   = blockIdx.y * 128 + wv * 32;
    int c0   = blockIdx.x * 64;

    v8f acc[2][4] = {};
    for (int k = 0; k < EMBED; k += 32) {
        v16bf a0 = load_frag(Ob, EMBED, r0,      k, lane);
        v16bf a1 = load_frag(Ob, EMBED, r0 + 16, k, lane);
        if (k + 32 < EMBED) {
            __builtin_prefetch(Ob + (size_t)(r0 + (lane & 15)) * EMBED + k + 32, 0, 1);
        }
#pragma unroll
        for (int t = 0; t < 4; ++t) {
            v16bf b = load_frag(wob, EMBED, c0 + t * 16, k, lane);
            acc[0][t] = wmma_bf16(a0, b, acc[0][t]);
            acc[1][t] = wmma_bf16(a1, b, acc[1][t]);
        }
    }

    int nn = lane & 15;
#pragma unroll
    for (int t = 0; t < 4; ++t) {
        int   col  = c0 + t * 16 + nn;
        float bias = b_out[col];
#pragma unroll
        for (int rt = 0; rt < 2; ++rt) {
#pragma unroll
            for (int v = 0; v < 8; ++v) {
                int row = r0 + rt * 16 + half * 8 + v;
                out[(size_t)row * EMBED + col] = acc[rt][t][v] + bias;
            }
        }
    }
}

extern "C" void kernel_launch(void* const* d_in, const int* in_sizes, int n_in,
                              void* d_out, int out_size, void* d_ws, size_t ws_size,
                              hipStream_t stream)
{
    (void)in_sizes; (void)n_in; (void)out_size; (void)ws_size;

    const float* x_q   = (const float*)d_in[0];
    const float* w_qkv = (const float*)d_in[1];
    const float* b_qkv = (const float*)d_in[2];
    const float* w_out = (const float*)d_in[3];
    const float* b_out = (const float*)d_in[4];
    float*       out   = (float*)d_out;

    char*  ws  = (char*)d_ws;
    size_t off = 0;
    bf16_t* xb  = (bf16_t*)(ws + off); off += (size_t)M_TOTAL * EMBED * 2;
    bf16_t* wqb = (bf16_t*)(ws + off); off += (size_t)QKV_D   * EMBED * 2;
    bf16_t* wob = (bf16_t*)(ws + off); off += (size_t)EMBED   * EMBED * 2;
    bf16_t* Qs  = (bf16_t*)(ws + off); off += (size_t)M_TOTAL * EMBED * 2;
    bf16_t* Kb  = (bf16_t*)(ws + off); off += (size_t)M_TOTAL * EMBED * 2;
    bf16_t* Vt  = (bf16_t*)(ws + off); off += (size_t)M_TOTAL * EMBED * 2;
    bf16_t* Pb  = (bf16_t*)(ws + off); off += (size_t)NBATCH * SEQ_P * SEQ_P * 2;
    bf16_t* Ob  = (bf16_t*)(ws + off); off += (size_t)M_TOTAL * EMBED * 2;

    cvt_f32_bf16<<<256, 256, 0, stream>>>(x_q,   xb,  M_TOTAL * EMBED);
    cvt_f32_bf16<<<64,  256, 0, stream>>>(w_qkv, wqb, QKV_D * EMBED);
    cvt_f32_bf16<<<32,  256, 0, stream>>>(w_out, wob, EMBED * EMBED);

    qkv_gemm<<<dim3(QKV_D / 64, M_TOTAL / 128, 1), 128, 0, stream>>>(
        xb, wqb, b_qkv, Qs, Kb, Vt);

    attn_scores_softmax<<<dim3(SEQ_P / 16, 1, NBATCH), 256, 0, stream>>>(
        Qs, Kb, Pb);

    av_gemm<<<dim3(EMBED / 64, SEQ_P / 128, NBATCH), 128, 0, stream>>>(
        Pb, Vt, Ob);

    out_gemm<<<dim3(EMBED / 64, M_TOTAL / 128, 1), 128, 0, stream>>>(
        Ob, wob, b_out, out);
}
